// TransformerEncoderLayer_78331613544930
// MI455X (gfx1250) — matrix operand order repaired and checked by reading, not compiled, not run on hardware
//
#include <hip/hip_runtime.h>
#include <hip/hip_bf16.h>

// ---------------- types ----------------
typedef __attribute__((ext_vector_type(16))) __bf16         v16bf;
typedef __attribute__((ext_vector_type(16))) unsigned short us16;
typedef __attribute__((ext_vector_type(8)))  unsigned short us8;
typedef __attribute__((ext_vector_type(4)))  unsigned short us4;
typedef __attribute__((ext_vector_type(8)))  float          v8f;

#define DMODEL 1024
#define NHEAD  16
#define HEAD_DIM 64
#define DFF    4096
#define BATCH  4
#define SEQ    1024
#define MROWS  (BATCH*SEQ)   // 4096
#define EPS    1e-5f

#define GEMM_BN 256   // block tile N
#define GEMM_BK 32    // k-step
#define LDS_LDK 40    // padded K-stride (elems) of the LDS B^T tile: 80 B, 16B-aligned rows

// ---------------- bf16 helpers (bit-level, no __bf16 arithmetic) ----------------
__device__ __forceinline__ unsigned short f2bf(float f) {
    unsigned u = __float_as_uint(f);
    u += 0x7FFFu + ((u >> 16) & 1u);   // round-to-nearest-even
    return (unsigned short)(u >> 16);
}

// ---------------- WMMA fragment loaders (contiguous k order k = 16*(lane/16) + s, same in A and B) ----------------
// A 16x32 bf16: lane&15 = M row; lanes 0-15 hold K {0..15}, lanes 16-31 hold K {16..31}
__device__ __forceinline__ v16bf load_a_frag(const unsigned short* p, int ld) {
    int lane = threadIdx.x & 31;
    const unsigned short* r = p + (size_t)(lane & 15) * ld + ((lane < 16) ? 0 : 16);
    us8 lo = *(const us8*)r;          // K run {0..7} / {16..23}
    us8 hi = *(const us8*)(r + 8);    // K run {8..15} / {24..31}
    us16 t;
#pragma unroll
    for (int j = 0; j < 8; ++j) { t[j] = lo[j]; t[8 + j] = hi[j]; }
    return __builtin_bit_cast(v16bf, t);
}
// B 32x16 from a transposed (N-major) view: element(k,n) at p[n*ld + k]; lanes 0-15 K=0..15,
// lanes 16-31 K=16..31, N = lane&15. Two contiguous b128 loads per lane.
__device__ __forceinline__ v16bf load_b_frag_T(const unsigned short* p, int ld) {
    int lane = threadIdx.x & 31;
    const unsigned short* r = p + (size_t)(lane & 15) * ld + ((lane < 16) ? 0 : 16);
    us8 lo = *(const us8*)r;
    us8 hi = *(const us8*)(r + 8);
    us16 t;
#pragma unroll
    for (int j = 0; j < 8; ++j) { t[j] = lo[j]; t[8 + j] = hi[j]; }
    return __builtin_bit_cast(v16bf, t);
}
// Same, from the padded LDS B^T tile (ld = LDS_LDK)
__device__ __forceinline__ v16bf load_b_frag_ldsT(const unsigned short* p) {
    int lane = threadIdx.x & 31;
    const unsigned short* r = p + (size_t)(lane & 15) * LDS_LDK + ((lane < 16) ? 0 : 16);
    us8 lo = *(const us8*)r;
    us8 hi = *(const us8*)(r + 8);
    us16 t;
#pragma unroll
    for (int j = 0; j < 8; ++j) { t[j] = lo[j]; t[8 + j] = hi[j]; }
    return __builtin_bit_cast(v16bf, t);
}
// A fragment from the packed-bf16 softmax rows in LDS.
// Packed row m lives at byte m*4096; keys [0,512) packed at bytes [0,1024),
// keys [512,1024) packed at bytes [2048,3072).
__device__ __forceinline__ v16bf load_a_frag_pk(const float* rowbase, int k0) {
    int lane = threadIdx.x & 31;
    const unsigned short* r = (const unsigned short*)((const char*)rowbase
                              + (size_t)(lane & 15) * (SEQ * 4)
                              + ((k0 >= SEQ / 2) ? 2048 : 0));
    int kk = (k0 & (SEQ / 2 - 1)) + ((lane < 16) ? 0 : 16);
    us8 lo = *(const us8*)(r + kk);
    us8 hi = *(const us8*)(r + kk + 8);
    us16 t;
#pragma unroll
    for (int j = 0; j < 8; ++j) { t[j] = lo[j]; t[8 + j] = hi[j]; }
    return __builtin_bit_cast(v16bf, t);
}
__device__ __forceinline__ v8f wmma_bf16(v16bf a, v16bf b, v8f c) {
    return __builtin_amdgcn_wmma_f32_16x16x32_bf16(false, a, false, b, (short)0, c, false, false);
}

// ---------------- elementwise / conversion kernels ----------------
__global__ void add_pos_cvt_kernel(const float* __restrict__ x, const float* __restrict__ pos,
                                   float* __restrict__ hf, unsigned short* __restrict__ hb, int n) {
    int i = blockIdx.x * blockDim.x + threadIdx.x;
    if (i < n) {
        float v = x[i] + pos[i];
        hf[i] = v;
        hb[i] = f2bf(v);
    }
}
// w: K x N (f32, row-major) -> wt: N x K (bf16), LDS-tiled so both sides stay coalesced
__global__ __launch_bounds__(256) void cvt_transpose_kernel(const float* __restrict__ w,
                                                            unsigned short* __restrict__ wt,
                                                            int K, int N) {
    __shared__ float tile[32][33];
    int tx = threadIdx.x & 31;
    int ty = threadIdx.x >> 5;       // 0..7
    int kbase = blockIdx.y * 32;
    int nbase = blockIdx.x * 32;
#pragma unroll
    for (int i = 0; i < 4; ++i)
        tile[ty + i * 8][tx] = w[(size_t)(kbase + ty + i * 8) * N + nbase + tx];
    __syncthreads();
#pragma unroll
    for (int i = 0; i < 4; ++i)
        wt[(size_t)(nbase + ty + i * 8) * K + kbase + tx] = f2bf(tile[tx][ty + i * 8]);
}

// ---------------- generic bf16 WMMA GEMM (weights pre-transposed: BT is N x K) ----------------
// C[M,N] = A[M,K] @ B[K,N], BT[n][k] = B[k][n].
// 256 threads = 8 waves, block tile 64(M) x 256(N), wave tile 32x64 (2x4 WMMA frags).
// BT tile (256 x 32 bf16, K contiguous) double-buffered in LDS via
// GLOBAL_LOAD_ASYNC_TO_LDS_B128 (ASYNCcnt), fragments read as 2x ds_load_b128.
template <bool HAS_BIAS, bool RELU, bool HAS_RES, bool OUT_F, bool OUT_BF, bool OUT_BF_T>
__global__ __launch_bounds__(256) void gemm_bf16_kernel(const unsigned short* __restrict__ A,
                                                        const unsigned short* __restrict__ BT,
                                                        const float* __restrict__ bias,
                                                        const float* __restrict__ residual,
                                                        float* __restrict__ outF,
                                                        unsigned short* __restrict__ outBf,
                                                        int M, int N, int K) {
    __shared__ unsigned short sB[2][GEMM_BN * LDS_LDK];   // 2 x 20 KB

    int tid  = threadIdx.x;
    int lane = tid & 31;
    int w    = tid >> 5;
    int wm   = w >> 2;          // 0..1
    int wn   = w & 3;           // 0..3
    int rowbase  = blockIdx.y * 64 + wm * 32;
    int colbase0 = blockIdx.x * GEMM_BN;
    int wcol     = wn * 64;

    // stage one 256(N) x 32(K) B^T tile (1024 x 16B chunks) into LDS buffer `buf`
    auto stage = [&](int buf, int kk) {
#pragma unroll
        for (int c = 0; c < 4; ++c) {
            int chunk = tid + c * 256;
            int nrow  = chunk >> 2;          // 0..255
            int kcol  = (chunk & 3) * 8;     // 0,8,16,24
            const unsigned short* gp = BT + (size_t)(colbase0 + nrow) * K + kk + kcol;
            unsigned ldsoff = (unsigned)(uintptr_t)&sB[buf][nrow * LDS_LDK + kcol];
            asm volatile("global_load_async_to_lds_b128 %0, %1, off"
                         :: "v"(ldsoff), "v"(gp)
                         : "memory");
        }
    };

    v8f acc[2][4] = {};
    const int nsteps = K / GEMM_BK;
    stage(0, 0);
    for (int s = 0; s < nsteps; ++s) {
        int kk = s * GEMM_BK;
        asm volatile("s_wait_asynccnt 0x0" ::: "memory");
        __syncthreads();
        if (s + 1 < nsteps) stage((s + 1) & 1, kk + GEMM_BK);

        const unsigned short* Abase = A + (size_t)rowbase * K + kk;
        __builtin_prefetch(Abase + (size_t)(lane & 15) * K + 2 * GEMM_BK, 0, 3);
        v16bf a0 = load_a_frag(Abase, K);
        v16bf a1 = load_a_frag(Abase + (size_t)16 * K, K);

        const unsigned short* bt = &sB[s & 1][wcol * LDS_LDK];
#pragma unroll
        for (int j = 0; j < 4; ++j) {
            v16bf b = load_b_frag_ldsT(bt + j * 16 * LDS_LDK);
            acc[0][j] = wmma_bf16(a0, b, acc[0][j]);
            acc[1][j] = wmma_bf16(a1, b, acc[1][j]);
        }
        __syncthreads();
    }

#pragma unroll
    for (int i = 0; i < 2; ++i)
#pragma unroll
        for (int j = 0; j < 4; ++j) {
            int col = colbase0 + wcol + j * 16 + (lane & 15);
            float bv = HAS_BIAS ? bias[col] : 0.f;
#pragma unroll
            for (int r = 0; r < 8; ++r) {
                int row = rowbase + i * 16 + r + ((lane < 16) ? 0 : 8);
                size_t idx = (size_t)row * N + col;
                float v = acc[i][j][r] + bv;
                if (RELU) v = v > 0.f ? v : 0.f;
                if (HAS_RES) v += residual[idx];
                if (OUT_F)    outF[idx] = v;
                if (OUT_BF)   outBf[idx] = f2bf(v);
                if (OUT_BF_T) outBf[(size_t)col * M + row] = f2bf(v);   // transposed store (V)
            }
        }
}

// ---------------- attention (per b,head,64-query block; 4 waves x 16 rows) ----------------
// dynamic LDS: 4 waves * 16 rows * 1024 keys * f32 (256 KB of the 320 KB WGP pool) + rowsums.
// Softmax packs exp() to bf16 in place so PV A-fragments are plain ds_load_b128.
// V is consumed transposed (Vt[D][B*S]) so PV B-fragments are contiguous b128 loads.
__global__ __launch_bounds__(128) void attention_kernel(const unsigned short* __restrict__ Qb,
                                                        const unsigned short* __restrict__ Kb,
                                                        const unsigned short* __restrict__ Vt,
                                                        const float* __restrict__ mask,
                                                        unsigned short* __restrict__ Ob) {
    extern __shared__ float smem[];
    int lane = threadIdx.x & 31;
    int wave = threadIdx.x >> 5;           // 0..3
    int qblk = blockIdx.x;
    int head = blockIdx.y;
    int b    = blockIdx.z;

    float* sw = smem + (size_t)wave * 16 * SEQ;
    float* rs = smem + (size_t)4 * 16 * SEQ;   // 64 rowsums

    int q0 = qblk * 64 + wave * 16;
    const unsigned short* Qp = Qb + ((size_t)(b * SEQ + q0)) * DMODEL + head * HEAD_DIM;
    v16bf a0 = load_a_frag(Qp, DMODEL);
    v16bf a1 = load_a_frag(Qp + 32, DMODEL);

    // ---- scores: S = Q K^T / 8 + additive mask -> LDS (f32) ----
    for (int t = 0; t < SEQ / 16; ++t) {
        const unsigned short* Kp = Kb + ((size_t)(b * SEQ + t * 16)) * DMODEL + head * HEAD_DIM;
        __builtin_prefetch(Kp + (size_t)(lane & 15) * DMODEL + 4 * DMODEL * 16, 0, 3);
        v16bf b0 = load_b_frag_T(Kp, DMODEL);        // d = 0..31
        v16bf b1 = load_b_frag_T(Kp + 32, DMODEL);   // d = 32..63
        v8f c = {};
        c = wmma_bf16(a0, b0, c);
        c = wmma_bf16(a1, b1, c);
        int col = t * 16 + (lane & 15);
        float madd = (mask[b * SEQ + col] - 1.0f) * 1e8f;
#pragma unroll
        for (int r = 0; r < 8; ++r) {
            int row = r + ((lane < 16) ? 0 : 8);
            sw[(size_t)row * SEQ + col] = c[r] * 0.125f + madd;
        }
    }
    __syncthreads();

    // ---- softmax: 2 lanes per row; pack exp() to bf16 in place (behind the read cursor) ----
    {
        int row  = lane & 15;
        int half = lane >> 4;
        float* rp = sw + (size_t)row * SEQ + half * (SEQ / 2);
        unsigned short* pp = (unsigned short*)((char*)(sw + (size_t)row * SEQ) + half * 2048);
        float mx = -3.0e38f;
        for (int j = 0; j < SEQ / 2; j += 4) {
            float4 v = *(const float4*)(rp + j);
            mx = fmaxf(mx, fmaxf(fmaxf(v.x, v.y), fmaxf(v.z, v.w)));
        }
        mx = fmaxf(mx, __shfl_xor(mx, 16, 32));
        float sum = 0.f;
        for (int j = 0; j < SEQ / 2; j += 4) {
            float4 v = *(const float4*)(rp + j);
            v.x = __expf(v.x - mx); v.y = __expf(v.y - mx);
            v.z = __expf(v.z - mx); v.w = __expf(v.w - mx);
            sum += v.x + v.y + v.z + v.w;
            us4 pk;
            pk[0] = f2bf(v.x); pk[1] = f2bf(v.y); pk[2] = f2bf(v.z); pk[3] = f2bf(v.w);
            *(us4*)(pp + j) = pk;    // bytes 2j..2j+7 <= read cursor 4j, own region only
        }
        sum += __shfl_xor(sum, 16, 32);
        rs[wave * 16 + row] = sum;
    }
    __syncthreads();

    // ---- PV: O = P V, contraction over 1024 keys in steps of 32 ----
    v8f o[4] = {};
    const unsigned short* Vh = Vt + (size_t)(head * HEAD_DIM) * MROWS + (size_t)b * SEQ;
    for (int ks = 0; ks < SEQ / 32; ++ks) {
        v16bf pa = load_a_frag_pk(sw, ks * 32);
        const unsigned short* Vp = Vh + ks * 32;
        __builtin_prefetch(Vp + (size_t)(lane & 15) * MROWS + 64, 0, 3);
#pragma unroll
        for (int n = 0; n < 4; ++n) {
            v16bf vb = load_b_frag_T(Vp + (size_t)(n * 16) * MROWS, MROWS);
            o[n] = wmma_bf16(pa, vb, o[n]);
        }
    }
    // ---- normalize by rowsum, write to (B,S,D) bf16 ----
#pragma unroll
    for (int n = 0; n < 4; ++n) {
#pragma unroll
        for (int r = 0; r < 8; ++r) {
            int row = r + ((lane < 16) ? 0 : 8);
            float inv = 1.0f / rs[wave * 16 + row];
            int s = q0 + row;
            int d = head * HEAD_DIM + n * 16 + (lane & 15);
            Ob[((size_t)(b * SEQ + s)) * DMODEL + d] = f2bf(o[n][r] * inv);
        }
    }
}

// ---------------- layernorm over last dim D=1024; one block per row ----------------
template <bool OUT_BF>
__global__ __launch_bounds__(256) void layernorm_kernel(const float* __restrict__ in,
                                                        const float* __restrict__ g,
                                                        const float* __restrict__ be,
                                                        float* __restrict__ outF,
                                                        unsigned short* __restrict__ outBf) {
    __shared__ float red1[256];
    __shared__ float red2[256];
    int row = blockIdx.x;
    int tid = threadIdx.x;
    const float* r = in + (size_t)row * DMODEL;
    float4 v = *(const float4*)(r + tid * 4);
    float s  = v.x + v.y + v.z + v.w;
    float sq = v.x * v.x + v.y * v.y + v.z * v.z + v.w * v.w;
    red1[tid] = s; red2[tid] = sq;
    __syncthreads();
    for (int off = 128; off > 0; off >>= 1) {
        if (tid < off) { red1[tid] += red1[tid + off]; red2[tid] += red2[tid + off]; }
        __syncthreads();
    }
    float mu  = red1[0] * (1.0f / DMODEL);
    float var = red2[0] * (1.0f / DMODEL) - mu * mu;
    float rstd = rsqrtf(var + EPS);
#pragma unroll
    for (int k = 0; k < 4; ++k) {
        int i = tid * 4 + k;
        float e = (k == 0) ? v.x : (k == 1) ? v.y : (k == 2) ? v.z : v.w;
        float y = (e - mu) * rstd * g[i] + be[i];
        outF[(size_t)row * DMODEL + i] = y;
        if (OUT_BF) outBf[(size_t)row * DMODEL + i] = f2bf(y);
    }
}

// ---------------- host-side launch ----------------
extern "C" void kernel_launch(void* const* d_in, const int* in_sizes, int n_in,
                              void* d_out, int out_size, void* d_ws, size_t ws_size,
                              hipStream_t stream) {
    (void)in_sizes; (void)n_in; (void)out_size; (void)ws_size;
    const float* x     = (const float*)d_in[0];
    const float* pos   = (const float*)d_in[1];
    const float* mask  = (const float*)d_in[2];
    const float* Wq    = (const float*)d_in[3];
    const float* Wk    = (const float*)d_in[4];
    const float* Wv    = (const float*)d_in[5];
    const float* Wo    = (const float*)d_in[6];
    const float* ln1_g = (const float*)d_in[7];
    const float* ln1_b = (const float*)d_in[8];
    const float* W1    = (const float*)d_in[9];
    const float* b1    = (const float*)d_in[10];
    const float* W2    = (const float*)d_in[11];
    const float* b2    = (const float*)d_in[12];
    const float* ln2_g = (const float*)d_in[13];
    const float* ln2_b = (const float*)d_in[14];

    char* wp = (char*)d_ws;
    auto carve = [&](size_t bytes) -> char* {
        char* p = wp;
        wp += (bytes + 255) & ~(size_t)255;
        return p;
    };
    const size_t MD = (size_t)MROWS * DMODEL;     // 4M
    const size_t MF = (size_t)MROWS * DFF;        // 16M
    const size_t DD = (size_t)DMODEL * DMODEL;    // 1M
    const size_t DF = (size_t)DMODEL * DFF;       // 4M

    float*          h_f32    = (float*)carve(MD * 4);
    unsigned short* h_bf     = (unsigned short*)carve(MD * 2);
    unsigned short* WqT      = (unsigned short*)carve(DD * 2);
    unsigned short* WkT      = (unsigned short*)carve(DD * 2);
    unsigned short* WvT      = (unsigned short*)carve(DD * 2);
    unsigned short* WoT      = (unsigned short*)carve(DD * 2);
    unsigned short* W1T      = (unsigned short*)carve(DF * 2);
    unsigned short* W2T      = (unsigned short*)carve(DF * 2);
    unsigned short* Qb       = (unsigned short*)carve(MD * 2);
    unsigned short* Kb       = (unsigned short*)carve(MD * 2);
    unsigned short* VtB      = (unsigned short*)carve(MD * 2);   // transposed V: [D][B*S]
    unsigned short* attn_bf  = (unsigned short*)carve(MD * 2);
    float*          mha_pre  = (float*)carve(MD * 4);
    float*          mha_f32  = (float*)carve(MD * 4);
    unsigned short* mha_bf   = (unsigned short*)carve(MD * 2);
    unsigned short* ffn1_bf  = (unsigned short*)carve(MF * 2);
    float*          ffn2_pre = (float*)carve(MD * 4);

    // 1) h = x + pos_em (f32 + bf16)
    add_pos_cvt_kernel<<<(MD + 255) / 256, 256, 0, stream>>>(x, pos, h_f32, h_bf, (int)MD);
    // 2) weight convert + transpose (bf16, N x K)
    dim3 gTdd(DMODEL / 32, DMODEL / 32);
    cvt_transpose_kernel<<<gTdd, 256, 0, stream>>>(Wq, WqT, DMODEL, DMODEL);
    cvt_transpose_kernel<<<gTdd, 256, 0, stream>>>(Wk, WkT, DMODEL, DMODEL);
    cvt_transpose_kernel<<<gTdd, 256, 0, stream>>>(Wv, WvT, DMODEL, DMODEL);
    cvt_transpose_kernel<<<gTdd, 256, 0, stream>>>(Wo, WoT, DMODEL, DMODEL);
    cvt_transpose_kernel<<<dim3(DFF / 32, DMODEL / 32), 256, 0, stream>>>(W1, W1T, DMODEL, DFF);
    cvt_transpose_kernel<<<dim3(DMODEL / 32, DFF / 32), 256, 0, stream>>>(W2, W2T, DFF, DMODEL);

    // 3) Q/K/V projections (Q,K row-major bf16; V transposed bf16)
    dim3 gD(DMODEL / GEMM_BN, MROWS / 64);   // (4, 64)
    gemm_bf16_kernel<false, false, false, false, true, false><<<gD, 256, 0, stream>>>(
        h_bf, WqT, nullptr, nullptr, nullptr, Qb, MROWS, DMODEL, DMODEL);
    gemm_bf16_kernel<false, false, false, false, true, false><<<gD, 256, 0, stream>>>(
        h_bf, WkT, nullptr, nullptr, nullptr, Kb, MROWS, DMODEL, DMODEL);
    gemm_bf16_kernel<false, false, false, false, false, true><<<gD, 256, 0, stream>>>(
        h_bf, WvT, nullptr, nullptr, nullptr, VtB, MROWS, DMODEL, DMODEL);

    // 4) attention
    dim3 gAtt(SEQ / 64, NHEAD, BATCH);
    size_t attSmem = ((size_t)4 * 16 * SEQ + 64) * sizeof(float);   // 256 KB + rowsums
    attention_kernel<<<gAtt, 128, attSmem, stream>>>(Qb, Kb, VtB, mask, attn_bf);

    // 5) output projection + residual -> LN1
    gemm_bf16_kernel<false, false, true, true, false, false><<<gD, 256, 0, stream>>>(
        attn_bf, WoT, nullptr, h_f32, mha_pre, nullptr, MROWS, DMODEL, DMODEL);
    layernorm_kernel<true><<<MROWS, 256, 0, stream>>>(mha_pre, ln1_g, ln1_b, mha_f32, mha_bf);

    // 6) FFN
    dim3 gF1(DFF / GEMM_BN, MROWS / 64);     // (16, 64)
    gemm_bf16_kernel<true, true, false, false, true, false><<<gF1, 256, 0, stream>>>(
        mha_bf, W1T, b1, nullptr, nullptr, ffn1_bf, MROWS, DFF, DMODEL);
    gemm_bf16_kernel<true, false, true, true, false, false><<<gD, 256, 0, stream>>>(
        ffn1_bf, W2T, b2, mha_f32, ffn2_pre, nullptr, MROWS, DMODEL, DFF);

    // 7) LN2 -> final output (f32)
    layernorm_kernel<false><<<MROWS, 256, 0, stream>>>(ffn2_pre, ln2_g, ln2_b, (float*)d_out, nullptr);
}
